// LSTM_171798692354
// MI455X (gfx1250) — matrix operand order repaired and checked by reading, not compiled, not run on hardware
//
#include <hip/hip_runtime.h>
#include <hip/hip_bf16.h>

// ---------------------------------------------------------------------------
// LSTM (T=512, B=128, I=512, H=1024) for MI455X (gfx1250, wave32, WMMA).
// Strategy:
//  * Pack x and weights to bf16 once (weights transposed to [N][K] so that
//    WMMA B-fragments are contiguous per-lane loads).
//  * Per timestep: one bf16 WMMA GEMM  [x_t | h_{t-1}] (128x1536) @ W^T
//    (1536x4096) -> gate preactivations (128x4096, f32 accum), then a
//    pointwise gate kernel. Weights (12MB bf16) stay L2-resident (192MB L2).
//  * t=0 is split into x-part / h-part GEMMs to reproduce the reference's
//    row-0 broadcast quirk on the c-gate.
// ---------------------------------------------------------------------------

typedef __attribute__((ext_vector_type(16))) __bf16 bf16x16;
typedef __attribute__((ext_vector_type(8)))  float  f32x8;

namespace lstm {
constexpr int T  = 512;
constexpr int B  = 128;
constexpr int I  = 512;
constexpr int H  = 1024;
constexpr int NG = 4 * H;     // 4096 (f,i,o,c concatenated)
constexpr int KT = I + H;     // 1536
constexpr int BH = B * H;     // 131072
}

union FragU {
    bf16x16 v;
    uint4   q[2];
};

// A fragment: 16x32 bf16 tile at (row0, kc) from row-major [rows x ld].
// ISA layout: lanes 0-15 -> M=lane, K = kc+{0..7, 16..23};
//             lanes 16-31 -> M=lane-16, K = kc+{8..15, 24..31}.
__device__ __forceinline__ bf16x16 load_a_frag(const __bf16* __restrict__ base,
                                               int ld, int row0, int kc, int lane) {
    const int r  = lane & 15;
    const int hi = lane >> 4;
    const __bf16* p = base + (size_t)(row0 + r) * ld + kc + hi * 8;
    FragU f;
    f.q[0] = *reinterpret_cast<const uint4*>(p);        // K chunk 0..7 (+hi*8)
    f.q[1] = *reinterpret_cast<const uint4*>(p + 16);   // K chunk 16..23 (+hi*8)
    return f.v;
}

// B fragment: 32x16 bf16 tile, weights stored transposed [N][K] row-major.
// ISA layout: lanes 0-15 -> N=lane, K = kc+{0..7, 16..23}; lanes 16-31 -> N=lane-16,
// K = kc+{8..15, 24..31}. Two 8-bf16 chunks per lane thanks to the [N][K] layout.
__device__ __forceinline__ bf16x16 load_b_frag(const __bf16* __restrict__ wt,
                                               int n0, int kc, int lane) {
    const int c  = lane & 15;
    const int hi = lane >> 4;
    const __bf16* p = wt + (size_t)(n0 + c) * lstm::KT + kc + hi * 8;
    FragU f;
    f.q[0] = *reinterpret_cast<const uint4*>(p);
    f.q[1] = *reinterpret_cast<const uint4*>(p + 16);
    return f.v;
}

// ---------------------------------------------------------------------------
// Step GEMM: out[B x NG] = [x_t | h] (B x KT) @ WT^T, K restricted to
// [kBegin,kEnd).  Block = 4 waves (128 thr), wave tile 32x64 (2x4 WMMA
// frags), block tile 64x128.  grid = (NG/128, B/64) = (32, 2).
// ---------------------------------------------------------------------------
__global__ __launch_bounds__(128)
void lstm_step_gemm(const __bf16* __restrict__ xpart,   // B x I  (timestep slice)
                    const __bf16* __restrict__ hpart,   // B x H
                    const __bf16* __restrict__ wt,      // NG x KT (transposed)
                    float* __restrict__ out,            // B x NG
                    int kBegin, int kEnd) {
    using namespace lstm;
    const int lane  = threadIdx.x & 31;
    const int wid   = threadIdx.x >> 5;
    const int waveM = wid >> 1;             // 0..1
    const int waveN = wid & 1;              // 0..1
    const int mBase = blockIdx.y * 64 + waveM * 32;
    const int nBase = blockIdx.x * 128 + waveN * 64;

    f32x8 acc[2][4];
#pragma unroll
    for (int mi = 0; mi < 2; ++mi)
#pragma unroll
        for (int ni = 0; ni < 4; ++ni)
            acc[mi][ni] = {};

    for (int k = kBegin; k < kEnd; k += 32) {
        const __bf16* abase;
        int lda, kk;
        if (k < I) { abase = xpart; lda = I; kk = k; }
        else       { abase = hpart; lda = H; kk = k - I; }

        // Prefetch next K-chunk of A into cache (global_prefetch_b8).
        if (k + 32 < kEnd && k + 32 >= I && k < I) {
            __builtin_prefetch(hpart + (size_t)(mBase + (lane & 15)) * H, 0, 0);
        }

        bf16x16 a0 = load_a_frag(abase, lda, mBase,      kk, lane);
        bf16x16 a1 = load_a_frag(abase, lda, mBase + 16, kk, lane);
        bf16x16 b0 = load_b_frag(wt, nBase,      k, lane);
        bf16x16 b1 = load_b_frag(wt, nBase + 16, k, lane);
        bf16x16 b2 = load_b_frag(wt, nBase + 32, k, lane);
        bf16x16 b3 = load_b_frag(wt, nBase + 48, k, lane);

        acc[0][0] = __builtin_amdgcn_wmma_f32_16x16x32_bf16(false, a0, false, b0, (short)0, acc[0][0], false, false);
        acc[0][1] = __builtin_amdgcn_wmma_f32_16x16x32_bf16(false, a0, false, b1, (short)0, acc[0][1], false, false);
        acc[0][2] = __builtin_amdgcn_wmma_f32_16x16x32_bf16(false, a0, false, b2, (short)0, acc[0][2], false, false);
        acc[0][3] = __builtin_amdgcn_wmma_f32_16x16x32_bf16(false, a0, false, b3, (short)0, acc[0][3], false, false);
        acc[1][0] = __builtin_amdgcn_wmma_f32_16x16x32_bf16(false, a1, false, b0, (short)0, acc[1][0], false, false);
        acc[1][1] = __builtin_amdgcn_wmma_f32_16x16x32_bf16(false, a1, false, b1, (short)0, acc[1][1], false, false);
        acc[1][2] = __builtin_amdgcn_wmma_f32_16x16x32_bf16(false, a1, false, b2, (short)0, acc[1][2], false, false);
        acc[1][3] = __builtin_amdgcn_wmma_f32_16x16x32_bf16(false, a1, false, b3, (short)0, acc[1][3], false, false);
    }

    // C layout: lanes 0-15 -> N=lane, M=v; lanes 16-31 -> N=lane-16, M=v+8.
    const int col = lane & 15;
    const int hi  = lane >> 4;
#pragma unroll
    for (int mi = 0; mi < 2; ++mi) {
#pragma unroll
        for (int ni = 0; ni < 4; ++ni) {
            const int row0 = mBase + mi * 16 + hi * 8;
            float* cp = out + (size_t)row0 * NG + nBase + ni * 16 + col;
#pragma unroll
            for (int v = 0; v < 8; ++v)
                cp[(size_t)v * NG] = acc[mi][ni][v];
        }
    }
}

// ---------------------------------------------------------------------------
// Pointwise gate kernel.  hg = fused preactivations (t>0) or x-part (t==0);
// hg2 = h-part at t==0 (row-0 broadcast quirk on the c-gate).
// ---------------------------------------------------------------------------
__global__ __launch_bounds__(256)
void lstm_pointwise(const float* __restrict__ hg,
                    const float* __restrict__ hg2,
                    const float* __restrict__ biasc,    // NG, (b*x + b*h)
                    float* __restrict__ cbuf,           // B x H (in/out)
                    __bf16* __restrict__ hbf,           // B x H (out, next A)
                    float* __restrict__ hseq_t,         // B x H (d_out slice)
                    float* __restrict__ hT,             // B x H or null
                    float* __restrict__ cT,             // B x H or null
                    int isT0) {
    using namespace lstm;
    const int idx = blockIdx.x * blockDim.x + threadIdx.x;
    if (idx >= BH) return;
    const int b = idx / H;
    const int j = idx - b * H;
    const size_t base = (size_t)b * NG;

    float zf = hg[base + j]         + biasc[j];
    float zi = hg[base + H + j]     + biasc[H + j];
    float zo = hg[base + 2 * H + j] + biasc[2 * H + j];
    float zc = hg[base + 3 * H + j] + biasc[3 * H + j];
    if (isT0) {
        zf += hg2[base + j];
        zi += hg2[base + H + j];
        zo += hg2[base + 2 * H + j];
        zc += hg2[(size_t)3 * H + j];   // row 0 of h-part, broadcast (quirk)
    }

    const float f = 1.0f / (1.0f + __expf(-zf));
    const float i = 1.0f / (1.0f + __expf(-zi));
    const float o = 1.0f / (1.0f + __expf(-zo));
    const float cn = f * cbuf[idx] + i * tanhf(zc);
    const float hn = o * tanhf(cn);

    cbuf[idx]   = cn;
    hbf[idx]    = (__bf16)hn;
    hseq_t[idx] = hn;
    if (hT) { hT[idx] = hn; cT[idx] = cn; }
}

// ---------------------------------------------------------------------------
// Packing kernels (run once per launch).
// ---------------------------------------------------------------------------
__global__ __launch_bounds__(256)
void pack_x_bf16(const float* __restrict__ x, __bf16* __restrict__ xb, long n) {
    long i = (long)blockIdx.x * blockDim.x + threadIdx.x;
    const long stride = (long)gridDim.x * blockDim.x;
    for (; i < n; i += stride) xb[i] = (__bf16)x[i];
}

__global__ __launch_bounds__(256)
void pack_weights(const float* __restrict__ Wfx, const float* __restrict__ Wix,
                  const float* __restrict__ Wox, const float* __restrict__ Wcx,
                  const float* __restrict__ Wfh, const float* __restrict__ Wih,
                  const float* __restrict__ Woh, const float* __restrict__ Wch,
                  const float* __restrict__ bfx, const float* __restrict__ bix,
                  const float* __restrict__ box, const float* __restrict__ bcx,
                  const float* __restrict__ bfh, const float* __restrict__ bih,
                  const float* __restrict__ boh, const float* __restrict__ bch,
                  __bf16* __restrict__ wt, float* __restrict__ biasc) {
    using namespace lstm;
    const long idx = (long)blockIdx.x * blockDim.x + threadIdx.x;
    const long total = (long)NG * KT;
    if (idx < total) {
        const int n = (int)(idx / KT);
        const int k = (int)(idx - (long)n * KT);
        const int g   = n >> 10;      // gate: 0=f 1=i 2=o 3=c
        const int col = n & (H - 1);
        const float* Wx[4] = {Wfx, Wix, Wox, Wcx};
        const float* Wh[4] = {Wfh, Wih, Woh, Wch};
        const float v = (k < I) ? Wx[g][(size_t)k * H + col]
                                : Wh[g][(size_t)(k - I) * H + col];
        wt[(size_t)n * KT + k] = (__bf16)v;   // transposed [N][K]
    }
    if (idx < NG) {
        const int g   = (int)(idx >> 10);
        const int col = (int)(idx & (H - 1));
        const float* bx[4] = {bfx, bix, box, bcx};
        const float* bh[4] = {bfh, bih, boh, bch};
        biasc[idx] = bx[g][col] + bh[g][col];
    }
}

__global__ __launch_bounds__(256)
void init_state(const float* __restrict__ h0, const float* __restrict__ c0,
                __bf16* __restrict__ hbf, float* __restrict__ cbuf) {
    using namespace lstm;
    const int idx = blockIdx.x * blockDim.x + threadIdx.x;
    if (idx >= BH) return;
    hbf[idx]  = (__bf16)h0[idx];   // h0 has leading dim 1 -> sum == h0[0]
    cbuf[idx] = c0[idx];
}

// ---------------------------------------------------------------------------
extern "C" void kernel_launch(void* const* d_in, const int* in_sizes, int n_in,
                              void* d_out, int out_size, void* d_ws, size_t ws_size,
                              hipStream_t stream) {
    using namespace lstm;
    const float* x   = (const float*)d_in[0];
    const float* h0  = (const float*)d_in[1];
    const float* c0  = (const float*)d_in[2];
    // params dict order: Wfx,bfx,Wix,bix,Wox,box,Wcx,bcx,Wfh,bfh,Wih,bih,Woh,boh,Wch,bch
    const float* Wfx = (const float*)d_in[3];
    const float* bfx = (const float*)d_in[4];
    const float* Wix = (const float*)d_in[5];
    const float* bix = (const float*)d_in[6];
    const float* Wox = (const float*)d_in[7];
    const float* box_ = (const float*)d_in[8];
    const float* Wcx = (const float*)d_in[9];
    const float* bcx = (const float*)d_in[10];
    const float* Wfh = (const float*)d_in[11];
    const float* bfh = (const float*)d_in[12];
    const float* Wih = (const float*)d_in[13];
    const float* bih = (const float*)d_in[14];
    const float* Woh = (const float*)d_in[15];
    const float* boh = (const float*)d_in[16];
    const float* Wch = (const float*)d_in[17];
    const float* bch = (const float*)d_in[18];

    float* out = (float*)d_out;   // [T*B*H | B*H (hT) | B*H (cT)]

    // Workspace carve-out (256B aligned slices).
    char* ws = (char*)d_ws;
    auto carve = [&](size_t bytes) -> char* {
        char* p = ws;
        ws += (bytes + 255) & ~(size_t)255;
        return p;
    };
    __bf16* wt    = (__bf16*)carve((size_t)NG * KT * sizeof(__bf16));   // 12 MB
    float*  biasc = (float*)carve((size_t)NG * sizeof(float));
    __bf16* xbf   = (__bf16*)carve((size_t)T * B * I * sizeof(__bf16)); // 64 MB
    __bf16* hbf   = (__bf16*)carve((size_t)BH * sizeof(__bf16));
    float*  cbuf  = (float*)carve((size_t)BH * sizeof(float));
    float*  hg    = (float*)carve((size_t)B * NG * sizeof(float));      // 2 MB
    float*  hg2   = (float*)carve((size_t)B * NG * sizeof(float));      // 2 MB

    // One-time packing.
    {
        const long nw = (long)NG * KT;
        pack_weights<<<(unsigned)((nw + 255) / 256), 256, 0, stream>>>(
            Wfx, Wix, Wox, Wcx, Wfh, Wih, Woh, Wch,
            bfx, bix, box_, bcx, bfh, bih, boh, bch, wt, biasc);
        pack_x_bf16<<<4096, 256, 0, stream>>>(x, xbf, (long)T * B * I);
        init_state<<<(BH + 255) / 256, 256, 0, stream>>>(h0, c0, hbf, cbuf);
    }

    const dim3 ggrid(NG / 128, B / 64);   // (32, 2)
    const unsigned pblocks = (BH + 255) / 256;

    for (int t = 0; t < T; ++t) {
        const __bf16* xt = xbf + (size_t)t * B * I;
        float* hseq_t = out + (size_t)t * BH;
        if (t == 0) {
            // Split x-part / h-part to honor the c-gate row-0 broadcast quirk.
            lstm_step_gemm<<<ggrid, 128, 0, stream>>>(xt, hbf, wt, hg, 0, I);
            lstm_step_gemm<<<ggrid, 128, 0, stream>>>(xt, hbf, wt, hg2, I, KT);
            lstm_pointwise<<<pblocks, 256, 0, stream>>>(
                hg, hg2, biasc, cbuf, hbf, hseq_t, nullptr, nullptr, 1);
        } else {
            lstm_step_gemm<<<ggrid, 128, 0, stream>>>(xt, hbf, wt, hg, 0, KT);
            float* hT = (t == T - 1) ? out + (size_t)T * BH : nullptr;
            float* cT = (t == T - 1) ? out + (size_t)T * BH + BH : nullptr;
            lstm_pointwise<<<pblocks, 256, 0, stream>>>(
                hg, hg, biasc, cbuf, hbf, hseq_t, hT, cT, 0);
        }
    }
}
